// CombineGraph_65025804861635
// MI455X (gfx1250) — matrix operand order repaired and checked by reading, not compiled, not run on hardware
//
#include <hip/hip_runtime.h>
#include <hip/hip_bf16.h>
#include <math.h>

#define B_    512
#define L_    50
#define NNEI_ 12
#define KATT_ 2
#define D_    128
#define V_    40000
#define ALPHA_ 0.2f
#define MU_    0.01f
#define INV_TEMP_ 10.0f
#define NEG_  (-9.0e15f)

typedef __attribute__((ext_vector_type(16))) _Float16     v16h;
typedef __attribute__((ext_vector_type(8)))  float        v8f;
typedef __attribute__((ext_vector_type(4)))  unsigned int v4u;
typedef __attribute__((ext_vector_type(8)))  int          v8i;
typedef __attribute__((ext_vector_type(4)))  int          v4i;

#if defined(__has_builtin)
#if __has_builtin(__builtin_amdgcn_tensor_load_to_lds) && __has_builtin(__builtin_amdgcn_s_wait_tensorcnt)
#define HAS_TDM 1
#endif
#endif
#ifndef HAS_TDM
#define HAS_TDM 0
#endif

__device__ __forceinline__ float sigmoidf_(float v) { return 1.0f / (1.0f + __expf(-v)); }
__device__ __forceinline__ float leakyf_(float v)   { return v > 0.0f ? v : ALPHA_ * v; }
__device__ __forceinline__ float wred_(float v) {
#pragma unroll
  for (int o = 16; o > 0; o >>= 1) v += __shfl_xor(v, o, 32);
  return v;
}
__device__ __forceinline__ float wmax_(float v) {
#pragma unroll
  for (int o = 16; o > 0; o >>= 1) v = fmaxf(v, __shfl_xor(v, o, 32));
  return v;
}

#if HAS_TDM
// Build a 2D TDM descriptor per CDNA5 ISA and issue
// tensor_load_to_lds (6-arg toolchain variant: g0,g1,g2,g3,g4,cpol).
// data_size = 4 bytes. Dims/strides in element units.
__device__ __forceinline__ void tdm_load_2d_f32(const float* gaddr, unsigned lds_off,
                                                unsigned tile_w, unsigned tile_h,
                                                unsigned tensor_w, unsigned tensor_h,
                                                unsigned stride_w) {
  const unsigned long long ga = (unsigned long long)(uintptr_t)gaddr;
  v4u g0;
  g0[0] = 1u;                                        // count=1, user mode
  g0[1] = lds_off;                                   // lds_addr (bytes)
  g0[2] = (unsigned)(ga & 0xFFFFFFFFu);              // global_addr[31:0]
  g0[3] = (unsigned)((ga >> 32) & 0x01FFFFFFu)       // global_addr[56:32]
          | (2u << 30);                              // type=2 ("image")
  v8i g1;
  g1[0] = (int)(2u << 16);                           // workgroup_mask=0, data_size=2 (4B)
  g1[1] = (int)((tensor_w & 0xFFFFu) << 16);         // tensor_dim0[15:0] @bits63:48
  g1[2] = (int)(((tensor_w >> 16) & 0xFFFFu)         // tensor_dim0[31:16]
          | ((tensor_h & 0xFFFFu) << 16));           // tensor_dim1[15:0]
  g1[3] = (int)(((tensor_h >> 16) & 0xFFFFu)         // tensor_dim1[31:16]
          | ((tile_w & 0xFFFFu) << 16));             // tile_dim0
  g1[4] = (int)(tile_h & 0xFFFFu);                   // tile_dim1 (tile_dim2=0)
  g1[5] = (int)stride_w;                             // tensor_dim0_stride[31:0]
  g1[6] = 0;                                         // stride hi / dim1_stride lo
  g1[7] = 0;
  v4i z4 = {0, 0, 0, 0};                             // 2D: groups 2/3 unused
  v8i z8 = {0, 0, 0, 0, 0, 0, 0, 0};
  __builtin_amdgcn_tensor_load_to_lds(g0, g1, z4, z4, z8, 0);
}
#endif

// ---------------------------------------------------------------------------
// NN WMMA GEMM, N=K=128 fixed (all weight GEMMs are DxD).
// B (128x128 f32, 64KB) is DMA'd to LDS once per block via the Tensor Data
// Mover; each wave owns a 16-row stripe and computes all 8 N-tiles, reusing
// one A fragment per K-step across 8 WMMAs.
// Fragment layouts used by this code (wave32):
//   A 16x32 : lane row = m0+(lane&15); half e -> K = e+16g
//   B 32x16 : lane col = n0+(lane&15); half e -> K = e+16g
//   C 16x16 : vgpr v   -> M = v+8g, N = lane&15
// ---------------------------------------------------------------------------
__global__ __launch_bounds__(128)
void gemm_nn_wmma(const float* __restrict__ A, const float* __restrict__ Bm,
                  float* __restrict__ C, int M, int accumulate) {
  __shared__ float sB[D_ * D_];                      // 64KB
  const int lane = threadIdx.x & 31;
  const int wave = threadIdx.x >> 5;
  const int g  = lane >> 4;
  const int lm = lane & 15;
#if HAS_TDM
  if (wave == 0) {                                   // wave-uniform branch
    tdm_load_2d_f32(Bm, (unsigned)(uintptr_t)(&sB[0]), D_, D_, D_, D_, D_);
    __builtin_amdgcn_s_wait_tensorcnt(0);
  }
#else
  for (int i = threadIdx.x; i < D_ * D_; i += 128) sB[i] = Bm[i];
#endif
  __syncthreads();

  const int m0 = (blockIdx.x * 4 + wave) * 16;
  v8f acc[8];
  if (accumulate) {
#pragma unroll
    for (int nt = 0; nt < 8; ++nt)
#pragma unroll
      for (int v = 0; v < 8; ++v)
        acc[nt][v] = C[(size_t)(m0 + v + 8 * g) * D_ + nt * 16 + lm];
  } else {
#pragma unroll
    for (int nt = 0; nt < 8; ++nt) acc[nt] = (v8f){};
  }

  for (int kk = 0; kk < D_; kk += 32) {
    v16h a;
    const float* Arow = A + (size_t)(m0 + lm) * D_ + kk + 16 * g;
#pragma unroll
    for (int e = 0; e < 8; ++e)  a[e] = (_Float16)Arow[e];
#pragma unroll
    for (int e = 8; e < 16; ++e) a[e] = (_Float16)Arow[e];
#pragma unroll
    for (int nt = 0; nt < 8; ++nt) {
      v16h b;
      const float* bp = &sB[(size_t)(kk + 16 * g) * D_ + nt * 16 + lm];
#pragma unroll
      for (int e = 0; e < 16; ++e) b[e] = (_Float16)bp[(size_t)e * D_];
      acc[nt] = __builtin_amdgcn_wmma_f32_16x16x32_f16(false, a, false, b,
                                                       (short)0, acc[nt], false, false);
    }
  }
#pragma unroll
  for (int nt = 0; nt < 8; ++nt)
#pragma unroll
    for (int v = 0; v < 8; ++v)
      C[(size_t)(m0 + v + 8 * g) * D_ + nt * 16 + lm] = acc[nt][v];
}

// ---------------------------------------------------------------------------
// scores = A[512,128] @ Bt[N,128]^T. Wave preloads all 4 A K-fragments into
// registers, then sweeps 8 N-subtiles (32 WMMAs per 128-column pass). Bt rows
// are contiguous in K (emb rows) -> per-lane b128 loads, L2-resident.
// ---------------------------------------------------------------------------
__global__ __launch_bounds__(128)
void gemm_nt_scores(const float* __restrict__ A, const float* __restrict__ Bt,
                    float* __restrict__ C, int N) {
  const int lane = threadIdx.x & 31;
  const int wave = threadIdx.x >> 5;
  const int g  = lane >> 4;
  const int lm = lane & 15;
  const int m0 = (blockIdx.y * 4 + wave) * 16;
  const int n0 = blockIdx.x * 128;

  v16h a[4];
  const float* Arow = A + (size_t)(m0 + lm) * D_;
#pragma unroll
  for (int k4 = 0; k4 < 4; ++k4) {
    const float* ap = Arow + k4 * 32 + 16 * g;
#pragma unroll
    for (int e = 0; e < 8; ++e)  a[k4][e] = (_Float16)ap[e];
#pragma unroll
    for (int e = 8; e < 16; ++e) a[k4][e] = (_Float16)ap[e];
  }

  for (int nt = 0; nt < 8; ++nt) {
    const int ncol = n0 + nt * 16 + lm;
    const int nld  = min(ncol, N - 1);               // clamp loads; no divergence
    const float* Brow = Bt + (size_t)nld * D_;
    v8f acc = {};
#pragma unroll
    for (int k4 = 0; k4 < 4; ++k4) {
      v16h b;
      const float* bp = Brow + k4 * 32 + 16 * g;
#pragma unroll
      for (int e = 0; e < 16; ++e) b[e] = (_Float16)bp[e];
      acc = __builtin_amdgcn_wmma_f32_16x16x32_f16(false, a[k4], false, b,
                                                   (short)0, acc, false, false);
    }
    if (ncol < N) {
#pragma unroll
      for (int v = 0; v < 8; ++v)
        C[(size_t)(m0 + v + 8 * g) * N + ncol] = acc[v];
    }
  }
}

// ---------------------------------------------------------------------------
// h[b,l,:] = embedding[inputs[b,l], :]
__global__ __launch_bounds__(128)
void k_embed(const int* __restrict__ inputs, const float* __restrict__ emb,
             float* __restrict__ h) {
  const int bl = blockIdx.x, d = threadIdx.x;
  h[(size_t)bl * D_ + d] = emb[(size_t)inputs[bl] * D_ + d];
}

// attribute aggregation: masked attention over NNEI neighbors, summed over K_att, /K_att
__global__ __launch_bounds__(32)
void k_attr(const float* __restrict__ emb, const float* __restrict__ h,
            const int* __restrict__ ids, const float* __restrict__ a_att,
            float* __restrict__ out) {
  const int bl = blockIdx.x, lane = threadIdx.x;
  float hv[4];
#pragma unroll
  for (int c = 0; c < 4; ++c) hv[c] = h[(size_t)bl * D_ + lane + 32 * c];
  float acc[4] = {0.f, 0.f, 0.f, 0.f};
  for (int k = 0; k < KATT_; ++k) {
    float ha[4];
#pragma unroll
    for (int c = 0; c < 4; ++c) ha[c] = hv[c] * a_att[k * D_ + lane + 32 * c];
    float e[NNEI_]; int idv[NNEI_];
    float emax = -3.0e38f;
    for (int n = 0; n < NNEI_; ++n) {
      const int id = ids[((size_t)k * B_ * L_ + bl) * NNEI_ + n];
      idv[n] = id;
      const float* nr = emb + (size_t)id * D_;
      float p = 0.f;
#pragma unroll
      for (int c = 0; c < 4; ++c) p += ha[c] * nr[lane + 32 * c];
      p = wred_(p);
      float ev = (id != 0) ? leakyf_(p) : NEG_;
      e[n] = ev;
      emax = fmaxf(emax, ev);
    }
    float esum = 0.f;
    for (int n = 0; n < NNEI_; ++n) { e[n] = __expf(e[n] - emax); esum += e[n]; }
    const float inv = 1.0f / esum;
    for (int n = 0; n < NNEI_; ++n) {
      const float wn = e[n] * inv;
      const float* nr = emb + (size_t)idv[n] * D_;
#pragma unroll
      for (int c = 0; c < 4; ++c) acc[c] += wn * nr[lane + 32 * c];
    }
  }
#pragma unroll
  for (int c = 0; c < 4; ++c)
    out[(size_t)bl * D_ + lane + 32 * c] = acc[c] * (1.0f / KATT_);
}

// relation-typed graph attention over L neighbors (one wave per (b,i))
__global__ __launch_bounds__(32)
void k_local(const float* __restrict__ h, const int* __restrict__ adj,
             const float* __restrict__ local_a, float* __restrict__ x) {
  const int b = blockIdx.x / L_, i = blockIdx.x % L_, lane = threadIdx.x;
  const float* hb = h + (size_t)b * L_ * D_;
  float hr[4][4];
#pragma unroll
  for (int c = 0; c < 4; ++c) {
    const float hi = hb[(size_t)i * D_ + lane + 32 * c];
#pragma unroll
    for (int r = 0; r < 4; ++r) hr[r][c] = hi * local_a[r * D_ + lane + 32 * c];
  }
  __shared__ float ev[L_];
  const int* arow = adj + ((size_t)b * L_ + i) * L_;
  for (int j = 0; j < L_; ++j) {
    float s0 = 0.f, s1 = 0.f, s2 = 0.f, s3 = 0.f;
#pragma unroll
    for (int c = 0; c < 4; ++c) {
      const float v = hb[(size_t)j * D_ + lane + 32 * c];
      s0 += hr[0][c] * v; s1 += hr[1][c] * v; s2 += hr[2][c] * v; s3 += hr[3][c] * v;
    }
    s0 = wred_(s0); s1 = wred_(s1); s2 = wred_(s2); s3 = wred_(s3);
    const int a = arow[j];
    float e = NEG_;
    if (a == 1) e = leakyf_(s0);
    else if (a == 2) e = leakyf_(s1);
    else if (a == 3) e = leakyf_(s2);
    else if (a == 4) e = leakyf_(s3);
    if (lane == 0) ev[j] = e;
  }
  __syncthreads();
  float m = -3.0e38f;
  for (int j = lane; j < L_; j += 32) m = fmaxf(m, ev[j]);
  m = wmax_(m);
  float ssum = 0.f;
  for (int j = lane; j < L_; j += 32) { const float p = __expf(ev[j] - m); ev[j] = p; ssum += p; }
  ssum = wred_(ssum);
  __syncthreads();
  const float inv = 1.0f / ssum;
  float acc[4] = {0.f, 0.f, 0.f, 0.f};
  for (int j = 0; j < L_; ++j) {
    const float p = ev[j] * inv;
#pragma unroll
    for (int c = 0; c < 4; ++c) acc[c] += p * hb[(size_t)j * D_ + lane + 32 * c];
  }
#pragma unroll
  for (int c = 0; c < 4; ++c)
    x[((size_t)b * L_ + i) * D_ + lane + 32 * c] = acc[c];
}

// gated mirror fusion: t = x@w1 + hf@w2 precomputed; updates x (in place) + mirror
__global__ __launch_bounds__(32)
void k_mirror(const float* __restrict__ t, const float* __restrict__ q,
              const float* __restrict__ hf, float* __restrict__ x,
              float* __restrict__ mir) {
  const int bl = blockIdx.x, lane = threadIdx.x;
  float s[4], p = 0.f;
#pragma unroll
  for (int c = 0; c < 4; ++c) {
    s[c] = tanhf(t[(size_t)bl * D_ + lane + 32 * c]);
    p += s[c] * q[lane + 32 * c];
  }
  const float beta = sigmoidf_(wred_(p));
#pragma unroll
  for (int c = 0; c < 4; ++c) {
    const size_t idx = (size_t)bl * D_ + lane + 32 * c;
    const float xv = x[idx], hv = hf[idx];
    x[idx]   = beta * hv + (1.0f - beta) * xv;
    mir[idx] = beta * xv + (1.0f - beta) * hv;
  }
}

// highway gate: x <- sigmoid(t)*h + (1-sigmoid(t))*x   (x becomes x_dot)
__global__ __launch_bounds__(128)
void k_highway(const float* __restrict__ t, const float* __restrict__ h,
               float* __restrict__ x) {
  const size_t i = (size_t)blockIdx.x * D_ + threadIdx.x;
  const float g = sigmoidf_(t[i]);
  x[i] = g * h[i] + (1.0f - g) * x[i];
}

// h_local[b,:] = sum_l mask[b,l] * x_dot[b,l,:]
__global__ __launch_bounds__(128)
void k_hlocal(const unsigned char* __restrict__ mask, const float* __restrict__ xd,
              float* __restrict__ hloc) {
  const int b = blockIdx.x, d = threadIdx.x;
  float v = 0.f;
  for (int l = 0; l < L_; ++l)
    if (mask[b * L_ + l]) v += xd[((size_t)b * L_ + l) * D_ + d];
  hloc[(size_t)b * D_ + d] = v;
}

// hp = x_dot + pos_embedding[l]   (written over the h buffer)
__global__ __launch_bounds__(128)
void k_hp(const float* __restrict__ xd, const float* __restrict__ pos,
          float* __restrict__ hp) {
  const int bl = blockIdx.x, d = threadIdx.x;
  hp[(size_t)bl * D_ + d] = xd[(size_t)bl * D_ + d] + pos[(bl % L_) * D_ + d];
}

// nh = sigmoid(t + bias[b] + glu3_b); beta2[b,l] = (nh . w) * (inputs!=0)
__global__ __launch_bounds__(32)
void k_nh_beta(const float* __restrict__ t, const float* __restrict__ bias,
               const float* __restrict__ g3b, const float* __restrict__ w,
               const int* __restrict__ inputs, float* __restrict__ beta2) {
  const int bl = blockIdx.x, b = bl / L_, lane = threadIdx.x;
  float p = 0.f;
#pragma unroll
  for (int c = 0; c < 4; ++c) {
    const int d = lane + 32 * c;
    const float nh = sigmoidf_(t[(size_t)bl * D_ + d] + bias[(size_t)b * D_ + d] + g3b[d]);
    p += nh * w[d];
  }
  p = wred_(p);
  if (lane == 0) beta2[bl] = (inputs[bl] != 0) ? p : 0.0f;
}

// zg[b,:] = sum_l beta2[b,l] * hp[b,l,:]
__global__ __launch_bounds__(128)
void k_zg(const float* __restrict__ beta2, const float* __restrict__ hp,
          float* __restrict__ zg) {
  const int b = blockIdx.x, d = threadIdx.x;
  float v = 0.f;
  for (int l = 0; l < L_; ++l)
    v += beta2[b * L_ + l] * hp[((size_t)b * L_ + l) * D_ + d];
  zg[(size_t)b * D_ + d] = v;
}

// zh = gf*h_local + (1-gf)*zg, gf = sigmoid(tg)*MU
__global__ __launch_bounds__(128)
void k_zh(const float* __restrict__ tg, const float* __restrict__ hloc,
          const float* __restrict__ zg, float* __restrict__ zh) {
  const size_t i = (size_t)blockIdx.x * D_ + threadIdx.x;
  const float gf = sigmoidf_(tg[i]) * MU_;
  zh[i] = gf * hloc[i] + (1.0f - gf) * zg[i];
}

// SSL loss: per (b,i) row softmax over j, masked -log, deterministic partials
__global__ __launch_bounds__(32)
void k_ssl(const float* __restrict__ hf1, const float* __restrict__ hf2,
           const int* __restrict__ simi, float* __restrict__ lossb) {
  const int b = blockIdx.x / L_, i = blockIdx.x % L_, lane = threadIdx.x;
  const float* f1 = hf1 + ((size_t)b * L_ + i) * D_;
  float a[4];
#pragma unroll
  for (int c = 0; c < 4; ++c) a[c] = f1[lane + 32 * c];
  __shared__ float sim[L_];
  for (int j = 0; j < L_; ++j) {
    const float* f2 = hf2 + ((size_t)b * L_ + j) * D_;
    float p = 0.f;
#pragma unroll
    for (int c = 0; c < 4; ++c) p += a[c] * f2[lane + 32 * c];
    p = wred_(p);
    if (lane == 0) sim[j] = p * INV_TEMP_;
  }
  __syncthreads();
  float m = -3.0e38f;
  for (int j = lane; j < L_; j += 32) m = fmaxf(m, sim[j]);
  m = wmax_(m);
  float ssum = 0.f;
  for (int j = lane; j < L_; j += 32) ssum += __expf(sim[j] - m);
  ssum = wred_(ssum);
  const int* srow = simi + ((size_t)b * L_ + i) * L_;
  float part = 0.f;
  for (int j = lane; j < L_; j += 32) {
    const float p = __expf(sim[j] - m) / ssum;
    if (srow[j] == 1) part += -__logf(p + 1e-8f);
  }
  part = wred_(part);
  if (lane == 0) lossb[blockIdx.x] = part;
}

__global__ __launch_bounds__(128)
void k_loss_reduce(const float* __restrict__ lossb, float* __restrict__ out0) {
  __shared__ float red[128];
  float v = 0.f;
  for (int i = threadIdx.x; i < B_ * L_; i += 128) v += lossb[i];
  red[threadIdx.x] = v;
  __syncthreads();
  for (int s = 64; s > 0; s >>= 1) {
    if (threadIdx.x < s) red[threadIdx.x] += red[threadIdx.x + s];
    __syncthreads();
  }
  if (threadIdx.x == 0) out0[0] = red[0] * (1.0f / B_);
}

// ---------------------------------------------------------------------------
extern "C" void kernel_launch(void* const* d_in, const int* in_sizes, int n_in,
                              void* d_out, int out_size, void* d_ws, size_t ws_size,
                              hipStream_t stream) {
  (void)in_sizes; (void)n_in; (void)out_size; (void)ws_size;
  const int*   inputs   = (const int*)d_in[0];
  const int*   adj      = (const int*)d_in[1];
  const unsigned char* lmask = (const unsigned char*)d_in[2];
  const int*   as_items = (const int*)d_in[3];
  const int*   as_ssl   = (const int*)d_in[4];
  const int*   simi     = (const int*)d_in[5];
  const float* emb      = (const float*)d_in[6];
  const float* pos      = (const float*)d_in[7];
  const float* w        = (const float*)d_in[8];
  const float* glu1_w   = (const float*)d_in[9];
  const float* glu2_w   = (const float*)d_in[10];
  const float* glu3_w   = (const float*)d_in[11];
  const float* glu3_b   = (const float*)d_in[12];
  const float* gate_w   = (const float*)d_in[13];
  const float* hw_w     = (const float*)d_in[14];
  const float* local_a  = (const float*)d_in[15];
  const float* mir_w1   = (const float*)d_in[16];
  const float* mir_w2   = (const float*)d_in[17];
  const float* mir_q    = (const float*)d_in[18];
  const float* a_att    = (const float*)d_in[19];

  float* out    = (float*)d_out;
  float* scores = out + 1;                      // [B, V-1] row-major

  const size_t BLD = (size_t)B_ * L_ * D_;
  float* ws    = (float*)d_ws;
  float* h     = ws + 0 * BLD;                  // later reused as hp
  float* hf1   = ws + 1 * BLD;
  float* hf2   = ws + 2 * BLD;
  float* x     = ws + 3 * BLD;                  // later x_new, then x_dot
  float* mir   = ws + 4 * BLD;
  float* t     = ws + 5 * BLD;                  // GEMM temp
  float* bias  = ws + 6 * BLD;                  // [B,D]
  float* hloc  = bias + (size_t)B_ * D_;
  float* zg    = hloc + (size_t)B_ * D_;
  float* tg    = zg   + (size_t)B_ * D_;
  float* zh    = tg   + (size_t)B_ * D_;
  float* beta2 = zh   + (size_t)B_ * D_;        // [B,L]
  float* lossb = beta2 + (size_t)B_ * L_;       // [B,L]

  const int BL = B_ * L_;
  const dim3 blk128(128), blk32(32);
  const dim3 gBL(BL), gB(B_);
  const dim3 gemmBL(BL / 64);                   // 25600 rows, 64 per block
  const dim3 gemmB (B_ / 64);                   // 512 rows

  // 1) embeddings + attribute attention + local graph attention
  k_embed<<<gBL, blk128, 0, stream>>>(inputs, emb, h);
  k_attr <<<gBL, blk32,  0, stream>>>(emb, h, as_items, a_att, hf1);
  k_attr <<<gBL, blk32,  0, stream>>>(emb, h, as_ssl,   a_att, hf2);
  k_local<<<gBL, blk32,  0, stream>>>(h, adj, local_a, x);

  // 2) mirror fusion: t = x@w1 + hf1@w2 ; update x, mir
  gemm_nn_wmma<<<gemmBL, blk128, 0, stream>>>(x,   mir_w1, t, BL, 0);
  gemm_nn_wmma<<<gemmBL, blk128, 0, stream>>>(hf1, mir_w2, t, BL, 1);
  k_mirror<<<gBL, blk32, 0, stream>>>(t, mir_q, hf1, x, mir);

  // 3) highway gate: t = h@W_top + x@W_bot ; x <- x_dot
  gemm_nn_wmma<<<gemmBL, blk128, 0, stream>>>(h, hw_w,           t, BL, 0);
  gemm_nn_wmma<<<gemmBL, blk128, 0, stream>>>(x, hw_w + D_ * D_, t, BL, 1);
  k_highway<<<gBL, blk128, 0, stream>>>(t, h, x);
  k_hlocal <<<gB,  blk128, 0, stream>>>(lmask, x, hloc);

  // 4) compute_score
  k_hp<<<gBL, blk128, 0, stream>>>(x, pos, h);  // h buffer now holds hp
  gemm_nn_wmma<<<gemmB,  blk128, 0, stream>>>(hloc, glu3_w, bias, B_, 0);
  gemm_nn_wmma<<<gemmBL, blk128, 0, stream>>>(h,    glu1_w, t,    BL, 0);
  gemm_nn_wmma<<<gemmBL, blk128, 0, stream>>>(mir,  glu2_w, t,    BL, 1);
  k_nh_beta<<<gBL, blk32, 0, stream>>>(t, bias, glu3_b, w, inputs, beta2);
  k_zg<<<gB, blk128, 0, stream>>>(beta2, h, zg);
  gemm_nn_wmma<<<gemmB, blk128, 0, stream>>>(zg,   gate_w,           tg, B_, 0);
  gemm_nn_wmma<<<gemmB, blk128, 0, stream>>>(hloc, gate_w + D_ * D_, tg, B_, 1);
  k_zh<<<gB, blk128, 0, stream>>>(tg, hloc, zg, zh);

  // 5) scores = zh @ emb[1:].T   (NT WMMA; emb rows contiguous in K)
  {
    const int Nout = V_ - 1;                    // 39999
    dim3 g((Nout + 127) / 128, B_ / 64);        // 313 x 8
    gemm_nt_scores<<<g, blk128, 0, stream>>>(zh, emb + D_, scores, Nout);
  }

  // 6) SSL similarity loss (deterministic two-stage reduction)
  k_ssl<<<gBL, blk32, 0, stream>>>(hf1, hf2, simi, lossb);
  k_loss_reduce<<<1, blk128, 0, stream>>>(lossb, out);
}
